// CoRNN_91122026152824
// MI455X (gfx1250) — matrix operand order repaired and checked by reading, not compiled, not run on hardware
//
#include <hip/hip_runtime.h>

// CoRNN on MI455X (gfx1250): persistent wave32 WMMA recurrence kernel.
// B=64, T=1024, I=256, H=512.  DT/GAMMA/EPS from reference.

#define DT_   0.054f
#define GAMMA_ 4.9f
#define EPS_  4.8f

constexpr int Bb = 64, Tt = 1024, Ii = 256, Hh = 512;
constexpr int WROW = Ii + 2 * Hh;          // 1280 (W row length)
constexpr int BH   = Bb * Hh;              // 32768 state elems
constexpr int NWG  = 32;                   // 4 batch-tiles x 8 h-slices

typedef __attribute__((ext_vector_type(16))) __bf16 v16bf;
typedef __attribute__((ext_vector_type(8)))  __bf16 v8bf;
typedef __attribute__((ext_vector_type(8)))  float  v8f;
typedef __attribute__((ext_vector_type(4)))  float  v4f;

// Build a 16-element bf16 WMMA fragment from two contiguous 8-element (16B) chunks.
__device__ __forceinline__ v16bf frag16(const __bf16* p0, const __bf16* p1) {
  v8bf lo = *(const v8bf*)p0;
  v8bf hi = *(const v8bf*)p1;
  v16bf r;
#pragma unroll
  for (int i = 0; i < 8; ++i) { r[i] = lo[i]; r[i + 8] = hi[i]; }
  return r;
}

// ---------------- init: W f32 -> bf16 (layout preserved), zero state + barrier counter
__global__ void cornn_init_kernel(const float* __restrict__ W,
                                  __bf16* __restrict__ Wbf,
                                  __bf16* __restrict__ state,
                                  unsigned* __restrict__ counter) {
  int tid = blockIdx.x * blockDim.x + threadIdx.x;
  int nth = gridDim.x * blockDim.x;
  for (int i = tid; i < Hh * WROW; i += nth) Wbf[i] = (__bf16)W[i];
  for (int i = tid; i < 4 * BH; i += nth)    state[i] = (__bf16)0.0f;
  if (tid == 0) *counter = 0u;
}

// ---------------- persistent recurrence kernel: 32 WGs x 128 threads (4 waves)
__global__ __launch_bounds__(128, 1)
void cornn_scan_kernel(const float* __restrict__ x,      // (B,T,I) f32
                       const float* __restrict__ bias,   // (H,)   f32
                       const __bf16* __restrict__ Wbf,   // (H, 1280) bf16
                       __bf16* __restrict__ state,       // [z0|z1|y0|y1] each B*H bf16
                       unsigned* __restrict__ counter,
                       float* __restrict__ outs,         // (B,T,H) f32
                       float* __restrict__ hout)         // (B,H)   f32
{
  extern __shared__ __bf16 lds[];
  __bf16* sWz = lds;                    // 64 x 512
  __bf16* sWy = sWz + 64 * 512;         // 64 x 512
  __bf16* sWx = sWy + 64 * 512;         // 64 x 256
  __bf16* sZ  = sWx + 64 * 256;         // 16 x 512 (hz, current)
  __bf16* sY  = sZ  + 16 * 512;         // 16 x 512 (hy, current)
  __bf16* sX  = sY  + 16 * 512;         // 16 x 256 (x_t, bf16)

  const int wg   = blockIdx.x;          // 0..31
  const int bt   = wg & 3;              // batch tile
  const int hs   = wg >> 2;             // 64-col h slice
  const int b0   = bt * 16;
  const int n0   = hs * 64;
  const int tid  = threadIdx.x;
  const int wave = tid >> 5;            // 0..3 -> 16-col subtile
  const int lane = tid & 31;
  const int lrow = lane & 15;
  const int hi16 = lane >> 4;           // 0 or 1

  // ---- load weight slices into LDS once (reused for all 1024 steps)
  for (int idx = tid; idx < 64 * 512 / 8; idx += 128) {
    int r = idx >> 6, c = (idx & 63) * 8;
    *(v8bf*)(sWz + r * 512 + c) = *(const v8bf*)(Wbf + (size_t)(n0 + r) * WROW + Ii + c);
    *(v8bf*)(sWy + r * 512 + c) = *(const v8bf*)(Wbf + (size_t)(n0 + r) * WROW + Ii + Hh + c);
  }
  for (int idx = tid; idx < 64 * 256 / 8; idx += 128) {
    int r = idx >> 5, c = (idx & 31) * 8;
    *(v8bf*)(sWx + r * 256 + c) = *(const v8bf*)(Wbf + (size_t)(n0 + r) * WROW + c);
  }

  const int ncol = n0 + wave * 16 + lrow;   // column this lane owns in C tiles
  const float bn = bias[ncol];

  // per-wave 16x16 state tiles in registers, WMMA C-layout: elem j -> row b0+hi16*8+j, col ncol
  v8f hyr, hzr;
#pragma unroll
  for (int j = 0; j < 8; ++j) { hyr[j] = 0.0f; hzr[j] = 0.0f; }

  const int aoff = hi16 * 8;    // A-fragment K base (lanes>=16 handle K+8 / K+24)
  const int boff = hi16 * 8;    // B-fragment K base (lanes>=16 handle K+8 / K+24)
  const int brow = wave * 16 + lrow;  // local weight row (= output column)

  for (int t = 0; t < Tt; ++t) {
    const int cur = t & 1, nxt = cur ^ 1;
    const __bf16* zc = state + (size_t)cur * BH;
    const __bf16* yc = state + (size_t)(2 + cur) * BH;

    // ---- stage current hz/hy rows (bf16) and x_t (f32->bf16) into LDS
    for (int idx = tid; idx < 16 * 512 / 8; idx += 128) {
      int r = idx >> 6, c = (idx & 63) * 8;
      *(v8bf*)(sZ + r * 512 + c) = *(const v8bf*)(zc + (size_t)(b0 + r) * Hh + c);
      *(v8bf*)(sY + r * 512 + c) = *(const v8bf*)(yc + (size_t)(b0 + r) * Hh + c);
    }
    for (int idx = tid; idx < 16 * 256 / 4; idx += 128) {
      int r = idx >> 6, c = (idx & 63) * 4;
      const float* xp = x + (size_t)(b0 + r) * Tt * Ii + (size_t)t * Ii + c;
      v4f xv = *(const v4f*)xp;
      __bf16* d = sX + r * 256 + c;
      d[0] = (__bf16)xv[0]; d[1] = (__bf16)xv[1];
      d[2] = (__bf16)xv[2]; d[3] = (__bf16)xv[3];
      if (t + 1 < Tt) __builtin_prefetch(xp + Ii, 0, 1);   // global_prefetch_b8
    }
    __syncthreads();

    // ---- pre = x_t@Wx^T + hz@Wz^T + hy@Wy^T  (f32 accumulate, bf16 WMMA)
    v8f acc;
#pragma unroll
    for (int j = 0; j < 8; ++j) acc[j] = 0.0f;

#pragma unroll
    for (int kc = 0; kc < 8; ++kc) {            // input term, K = 256
      int k0 = kc * 32;
      v16bf a  = frag16(sX + lrow * 256 + k0 + aoff, sX + lrow * 256 + k0 + aoff + 16);
      v16bf bm = frag16(sWx + brow * 256 + k0 + boff, sWx + brow * 256 + k0 + boff + 16);
      acc = __builtin_amdgcn_wmma_f32_16x16x32_bf16(false, a, false, bm, (short)0, acc, false, false);
    }
#pragma unroll
    for (int kc = 0; kc < 16; ++kc) {           // recurrent terms, K = 512
      int k0 = kc * 32;
      v16bf az = frag16(sZ + lrow * 512 + k0 + aoff, sZ + lrow * 512 + k0 + aoff + 16);
      v16bf bz = frag16(sWz + brow * 512 + k0 + boff, sWz + brow * 512 + k0 + boff + 16);
      acc = __builtin_amdgcn_wmma_f32_16x16x32_bf16(false, az, false, bz, (short)0, acc, false, false);
      v16bf ay = frag16(sY + lrow * 512 + k0 + aoff, sY + lrow * 512 + k0 + aoff + 16);
      v16bf by = frag16(sWy + brow * 512 + k0 + boff, sWy + brow * 512 + k0 + boff + 16);
      acc = __builtin_amdgcn_wmma_f32_16x16x32_bf16(false, ay, false, by, (short)0, acc, false, false);
    }

    // ---- elementwise update (register-local state), publish bf16 state + f32 outs
    __bf16* zn = state + (size_t)nxt * BH;
    __bf16* yn = state + (size_t)(2 + nxt) * BH;
    const int rbase = b0 + hi16 * 8;
#pragma unroll
    for (int j = 0; j < 8; ++j) {
      float p  = acc[j] + bn;
      float z  = hzr[j], y = hyr[j];
      float zv = z + DT_ * (tanhf(p) - GAMMA_ * y - EPS_ * z);
      float yv = y + DT_ * zv;
      hzr[j] = zv; hyr[j] = yv;
      int rg = rbase + j;
      outs[((size_t)rg * Tt + t) * Hh + ncol] = yv;
      zn[(size_t)rg * Hh + ncol] = (__bf16)zv;
      yn[(size_t)rg * Hh + ncol] = (__bf16)yv;
    }

    // ---- device-wide step barrier (32 persistent WGs)
    __threadfence();
    __syncthreads();
    if (tid == 0) {
      atomicAdd(counter, 1u);
      const unsigned target = (unsigned)NWG * (unsigned)(t + 1);
      volatile unsigned* vc = counter;
      while (*vc < target) __builtin_amdgcn_s_sleep(2);
    }
    __syncthreads();
    __threadfence();   // acquire: invalidate stale state lines before next stage
  }

  // ---- final hy -> h_out
#pragma unroll
  for (int j = 0; j < 8; ++j)
    hout[(size_t)(b0 + hi16 * 8 + j) * Hh + ncol] = hyr[j];
}

extern "C" void kernel_launch(void* const* d_in, const int* in_sizes, int n_in,
                              void* d_out, int out_size, void* d_ws, size_t ws_size,
                              hipStream_t stream) {
  const float* x  = (const float*)d_in[0];   // (B,T,I)
  const float* W  = (const float*)d_in[1];   // (H, I+2H)
  const float* bv = (const float*)d_in[2];   // (H,)

  float* outs = (float*)d_out;                      // (B,T,H)
  float* hout = outs + (size_t)Bb * Tt * Hh;        // (1,B,H)

  // workspace layout: [counter 256B][Wbf bf16 H*1280][state bf16 4*B*H]
  uint8_t* ws = (uint8_t*)d_ws;
  unsigned* counter = (unsigned*)ws;
  __bf16* Wbf   = (__bf16*)(ws + 256);
  __bf16* state = (__bf16*)(ws + 256 + (size_t)Hh * WROW * sizeof(__bf16));

  cornn_init_kernel<<<256, 256, 0, stream>>>(W, Wbf, state, counter);

  constexpr size_t LDS_BYTES =
      (size_t)(64 * 512 * 2 + 64 * 256 + 16 * 512 * 2 + 16 * 256) * sizeof(__bf16); // 204800
  cornn_scan_kernel<<<NWG, 128, LDS_BYTES, stream>>>(x, bv, Wbf, state, counter, outs, hout);
}
